// EdgeEnhancement_1065151889572
// MI455X (gfx1250) — matrix operand order repaired and checked by reading, not compiled, not run on hardware
//
#include <hip/hip_runtime.h>
#include <hip/hip_bf16.h>
#include <math.h>

// ---------------------------------------------------------------------------
// EdgeEnhancement for MI455X (gfx1250, wave32, WMMA).
// 1x1 convs -> bf16 WMMA GEMMs (v_wmma_f32_16x16x32_bf16, f32 accum).
// Activations for GEMMs are kept in NHWC bf16 so a B-fragment is ONE 32-byte
// vector load per lane (vs 16 strided u16 gathers). Bias+BN folded to
// scale/shift, epilogue fully vectorized (v8f loads, v8bf 16B stores).
// Stencil kernels (gates / 15x15 ctx / pool / deform) are VALU; x and all
// intermediates are L2-resident (192MB L2), so re-reads stay on-chip.
// Workspace: ~135MB.
// ---------------------------------------------------------------------------

typedef __bf16 bf16_t;
typedef __attribute__((ext_vector_type(16))) __bf16 v16bf;
typedef __attribute__((ext_vector_type(8)))  __bf16 v8bf;
typedef __attribute__((ext_vector_type(8)))  float  v8f;

#define HW_   16384   // H*W
#define W_    128
#define H_    128
#define C_    128
#define B_    4
#define EPS_  1e-5f

__device__ __forceinline__ float sigmoidf_(float t) { return 1.f / (1.f + __expf(-t)); }
__device__ __forceinline__ float bnf_(float v, float g, float b, float m, float var) {
  return (v - m) * (g * __frsqrt_rn(var + EPS_)) + b;
}

// ---------------------------------------------------------------------------
// Pack a Cout x Cin fp32 weight matrix into bf16 WMMA A-fragment layout.
// Contiguous k order used here (same as B, not AMD's table order): lane<16 holds row m=lane, K=0..15;
// lane>=16 holds row m=lane-16, K=16..31; 2 bf16 per VGPR.
// Packed index = ((mt*Kt + kt)*32 + lane)*16 + j  (16 bf16 = 32B per lane).
// ---------------------------------------------------------------------------
__global__ void pack_w_kernel(const float* __restrict__ src, bf16_t* __restrict__ dst,
                              int M, int K, int Mpad) {
  int idx = blockIdx.x * blockDim.x + threadIdx.x;
  int total = Mpad * K;
  if (idx >= total) return;
  int j    = idx & 15;
  int lane = (idx >> 4) & 31;
  int tile = idx >> 9;
  int Kt = K >> 5;
  int kt = tile % Kt;
  int mt = tile / Kt;
  int m  = mt * 16 + (lane & 15);
  int base = (lane & 16) ? 16 : 0;
  int kl = base + j;
  int k  = kt * 32 + kl;
  float v = (m < M) ? src[m * K + k] : 0.f;
  dst[idx] = (bf16_t)v;
}

// Fold conv-bias + BN into y = acc*scale[m] + shift[m].
__global__ void bnfold_kernel(const float* __restrict__ bias, const float* __restrict__ g,
                              const float* __restrict__ be, const float* __restrict__ m,
                              const float* __restrict__ v, int M, int Mpad,
                              float* __restrict__ scale, float* __restrict__ shift) {
  int i = blockIdx.x * blockDim.x + threadIdx.x;
  if (i >= Mpad) return;
  if (i < M) {
    float sc = g[i] * __frsqrt_rn(v[i] + EPS_);
    scale[i] = sc;
    shift[i] = (bias[i] - m[i]) * sc + be[i];
  } else {
    scale[i] = 0.f;
    shift[i] = 0.f;
  }
}

// NCHW f32 -> NHWC bf16 copy of x (for GEMM epilogue gating multiplies).
__global__ void to_nhwc_kernel(const float* __restrict__ x, bf16_t* __restrict__ xnh) {
  int idx = blockIdx.x * 256 + threadIdx.x;     // bc*HW + s
  int bc = idx >> 14;
  int s  = idx & (HW_ - 1);
  int p  = ((bc >> 7) << 14) + s;               // b*HW + s
  xnh[((size_t)p << 7) + (bc & (C_ - 1))] = (bf16_t)x[idx];
}

// ---------------------------------------------------------------------------
// Fused gate kernel: o_sum = sum_i sigmoid(bn(dwconv_i(x)))*x, for the
// 1x15 / 15x1 / 3x3 depthwise branches. One block = one (b,c,row).
// Output NHWC bf16.
// ---------------------------------------------------------------------------
__global__ void gates_kernel(const float* __restrict__ x,
    const float* __restrict__ w1, const float* __restrict__ b1, const float* __restrict__ g1,
    const float* __restrict__ be1, const float* __restrict__ m1, const float* __restrict__ v1,
    const float* __restrict__ w2, const float* __restrict__ b2, const float* __restrict__ g2,
    const float* __restrict__ be2, const float* __restrict__ m2, const float* __restrict__ v2,
    const float* __restrict__ w3, const float* __restrict__ b3, const float* __restrict__ g3,
    const float* __restrict__ be3, const float* __restrict__ m3, const float* __restrict__ v3,
    bf16_t* __restrict__ osum) {
  __shared__ float rowbuf[W_];
  const int tid = threadIdx.x;            // x coordinate, 0..127
  const int yy  = blockIdx.x % H_;
  const int bc  = blockIdx.x / H_;        // b*C + c
  const int c   = bc & (C_ - 1);
  const float* plane = x + ((size_t)bc << 14);
  const float center = plane[yy * W_ + tid];
  rowbuf[tid] = center;
  __syncthreads();

  float s1 = b1[c];
  #pragma unroll
  for (int t = 0; t < 15; ++t) {
    int xx = tid + t - 7;
    if (xx >= 0 && xx < W_) s1 += rowbuf[xx] * w1[c * 15 + t];
  }
  float s2 = b2[c];
  #pragma unroll
  for (int t = 0; t < 15; ++t) {
    int y2 = yy + t - 7;
    if (y2 >= 0 && y2 < H_) s2 += plane[y2 * W_ + tid] * w2[c * 15 + t];
  }
  float s3 = b3[c];
  #pragma unroll
  for (int i = 0; i < 3; ++i) {
    int y2 = yy - 1 + i;
    if (y2 < 0 || y2 >= H_) continue;
    #pragma unroll
    for (int j = 0; j < 3; ++j) {
      int xx = tid - 1 + j;
      if (xx < 0 || xx >= W_) continue;
      s3 += plane[y2 * W_ + xx] * w3[c * 9 + i * 3 + j];
    }
  }
  float gsum = sigmoidf_(bnf_(s1, g1[c], be1[c], m1[c], v1[c])) * center
             + sigmoidf_(bnf_(s2, g2[c], be2[c], m2[c], v2[c])) * center
             + sigmoidf_(bnf_(s3, g3[c], be3[c], m3[c], v3[c])) * center;
  const int p = ((bc >> 7) << 14) + yy * W_ + tid;   // b*HW + s
  osum[((size_t)p << 7) + c] = (bf16_t)gsum;
}

// ---------------------------------------------------------------------------
// WMMA GEMM for 1x1 convs over NHWC bf16 activations.
// One wave: 16-pixel column x MT Cout-tiles; B frag = one 32B vector load.
// MODE: 0 = store f32 offsets NHWC (Mvalid=18)
//       1 = *x -> bf16 NHWC (dcb)    2 = sigmoid*x -> bf16 NHWC (ns)
//       3 = store f32 NCHW (final output)
// ---------------------------------------------------------------------------
template <int MT, int KT, int MODE>
__global__ void gemm1x1_kernel(const bf16_t* __restrict__ Apk,
                               const bf16_t* __restrict__ B1,
                               const bf16_t* __restrict__ B2,
                               int Ksplit, int Mvalid,
                               const float* __restrict__ scale, const float* __restrict__ shift,
                               const bf16_t* __restrict__ xnh,
                               float* __restrict__ outF, bf16_t* __restrict__ outH) {
  const int lane  = threadIdx.x & 31;
  const int ntile = blockIdx.x * (blockDim.x >> 5) + (threadIdx.x >> 5);
  const int npix  = ntile * 16 + (lane & 15);

  v8f acc[MT];
  #pragma unroll
  for (int mt = 0; mt < MT; ++mt)
    #pragma unroll
    for (int i = 0; i < 8; ++i) acc[mt][i] = 0.f;

  const int hi = (lane & 16) ? 16 : 0;     // B frag: lanes 16-31 carry K+16..K+31
  const size_t prow = ((size_t)npix) << 7; // NHWC row base (128 channels)
  #pragma unroll
  for (int kt = 0; kt < KT; ++kt) {
    const int kbase = kt * 32 + hi;
    const bf16_t* bp = (kbase < Ksplit) ? (B1 + prow + kbase)
                                        : (B2 + prow + (kbase - Ksplit));
    const v16bf bfrag = *reinterpret_cast<const v16bf*>(bp);
    #pragma unroll
    for (int mt = 0; mt < MT; ++mt) {
      const v16bf afrag = *reinterpret_cast<const v16bf*>(
          Apk + (((size_t)((mt * KT + kt) * 32 + lane)) << 4));
      acc[mt] = __builtin_amdgcn_wmma_f32_16x16x32_bf16(
          false, afrag, false, bfrag, (short)0, acc[mt], false, false);
    }
  }

  const int rof = (lane & 16) ? 8 : 0;     // C/D frag: lanes 16-31 hold M=8..15
  const int b = npix >> 14;
  const int s = npix & (HW_ - 1);
  #pragma unroll
  for (int mt = 0; mt < MT; ++mt) {
    const int m0 = mt * 16 + rof;
    const v8f sc = *reinterpret_cast<const v8f*>(scale + m0);
    const v8f sh = *reinterpret_cast<const v8f*>(shift + m0);
    if (MODE == 0) {
      #pragma unroll
      for (int r = 0; r < 8; ++r) {
        int m = m0 + r;
        if (m < Mvalid) outF[(size_t)npix * 18 + m] = acc[mt][r] * sc[r] + sh[r];
      }
    } else if (MODE == 3) {
      #pragma unroll
      for (int r = 0; r < 8; ++r) {
        outF[(((size_t)(b * C_ + m0 + r)) << 14) + s] = acc[mt][r] * sc[r] + sh[r];
      }
    } else {
      const v8bf xv = *reinterpret_cast<const v8bf*>(xnh + prow + m0);
      v8bf res;
      #pragma unroll
      for (int r = 0; r < 8; ++r) {
        float y = acc[mt][r] * sc[r] + sh[r];
        if (MODE == 2) y = sigmoidf_(y);
        res[r] = (bf16_t)(y * (float)xv[r]);
      }
      *reinterpret_cast<v8bf*>(outH + prow + m0) = res;  // one 16B store
    }
  }
}

// ---------------------------------------------------------------------------
// Deformable sampling + per-channel 3x3 reduction (einsum over 9 taps).
// offsets NHWC f32 (B,HW,18) -> contiguous per-pixel reads.
// Output dout NHWC bf16.
// ---------------------------------------------------------------------------
__global__ void deform_kernel(const float* __restrict__ x, const float* __restrict__ offs,
                              const float* __restrict__ dw, bf16_t* __restrict__ dout) {
  const int idx = blockIdx.x * 256 + threadIdx.x;     // b*C*HW + c*HW + s
  const int b = idx >> 21;
  const int c = (idx >> 14) & (C_ - 1);
  const int s = idx & (HW_ - 1);
  const int yy = s >> 7, xx = s & (W_ - 1);
  const int p = (b << 14) + s;                        // pixel index
  const float* plane = x + ((size_t)(b * C_ + c) << 14);
  const float* ob = offs + (size_t)p * 18;
  float accv = 0.f;
  #pragma unroll
  for (int k = 0; k < 9; ++k) {
    const int ki = k / 3, kj = k % 3;
    float dy = ob[2 * k];
    float dx = ob[2 * k + 1];
    float py = (float)(yy - 1 + ki) + dy;
    float px = (float)(xx - 1 + kj) + dx;
    float y0f = floorf(py), x0f = floorf(px);
    float ly = py - y0f, lx = px - x0f;
    int y0 = (int)y0f, x0 = (int)x0f;
    float v = 0.f;
    #pragma unroll
    for (int di = 0; di < 2; ++di) {
      #pragma unroll
      for (int dj = 0; dj < 2; ++dj) {
        int yc = y0 + di, xc = x0 + dj;
        float wgt = (di ? ly : 1.f - ly) * (dj ? lx : 1.f - lx);
        if (yc >= 0 && yc < H_ && xc >= 0 && xc < W_) v += plane[yc * W_ + xc] * wgt;
      }
    }
    accv += v * dw[c * 9 + k];
  }
  dout[((size_t)p << 7) + c] = (bf16_t)accv;
}

// ---------------------------------------------------------------------------
// 15x15 depthwise conv (ns_ctx) + BN. LDS-tiled: 16x16 outputs per block,
// 30x30 halo tile + 225 weights in LDS. Output NCHW f32 (pooling input).
// ---------------------------------------------------------------------------
__global__ void nsctx_kernel(const float* __restrict__ x, const float* __restrict__ w,
                             const float* __restrict__ bias, const float* __restrict__ g,
                             const float* __restrict__ be, const float* __restrict__ m,
                             const float* __restrict__ v, float* __restrict__ xcon) {
  __shared__ float tile[30][30];
  __shared__ float wt[225];
  const int tid = threadIdx.x;        // 256 threads
  const int blk = blockIdx.x;
  const int tx = blk & 7, ty = (blk >> 3) & 7;
  const int bc = blk >> 6;            // b*C + c
  const int c  = bc & (C_ - 1);
  const float* plane = x + ((size_t)bc << 14);
  const int y0 = ty * 16 - 7, x0 = tx * 16 - 7;
  for (int i = tid; i < 900; i += 256) {
    int iy = i / 30, ix = i % 30;
    int yy = y0 + iy, xx = x0 + ix;
    tile[iy][ix] = (yy >= 0 && yy < H_ && xx >= 0 && xx < W_) ? plane[yy * W_ + xx] : 0.f;
  }
  for (int i = tid; i < 225; i += 256) wt[i] = w[c * 225 + i];
  __syncthreads();
  const int ly = tid >> 4, lx = tid & 15;
  float sum = bias[c];
  for (int ki = 0; ki < 15; ++ki)
    #pragma unroll
    for (int kj = 0; kj < 15; ++kj)
      sum += tile[ly + ki][lx + kj] * wt[ki * 15 + kj];
  xcon[((size_t)bc << 14) + (ty * 16 + ly) * W_ + (tx * 16 + lx)] =
      bnf_(sum, g[c], be[c], m[c], v[c]);
}

// ---------------------------------------------------------------------------
// diff = xcon - boxavg3x3(xcon)  (zero-padded sum / 9) -> NHWC bf16.
// ---------------------------------------------------------------------------
__global__ void pooldiff_kernel(const float* __restrict__ xcon, bf16_t* __restrict__ diff) {
  const int idx = blockIdx.x * 256 + threadIdx.x;   // bc*HW + s
  const int bc = idx >> 14;
  const int s  = idx & (HW_ - 1);
  const int yy = s >> 7, xx = s & (W_ - 1);
  const float* plane = xcon + ((size_t)bc << 14);
  float sum = 0.f;
  #pragma unroll
  for (int i = 0; i < 3; ++i) {
    int yc = yy - 1 + i;
    if (yc < 0 || yc >= H_) continue;
    #pragma unroll
    for (int j = 0; j < 3; ++j) {
      int xc = xx - 1 + j;
      if (xc < 0 || xc >= W_) continue;
      sum += plane[yc * W_ + xc];
    }
  }
  const int p = ((bc >> 7) << 14) + s;
  diff[((size_t)p << 7) + (bc & (C_ - 1))] = (bf16_t)(plane[s] - sum * (1.f / 9.f));
}

// ---------------------------------------------------------------------------
extern "C" void kernel_launch(void* const* d_in, const int* in_sizes, int n_in,
                              void* d_out, int out_size, void* d_ws, size_t ws_size,
                              hipStream_t stream) {
  (void)in_sizes; (void)n_in; (void)out_size; (void)ws_size;
  const float* x = (const float*)d_in[0];
  // Flattened param order (setup_inputs dict insertion order): branch={w,b,bn{g,b,m,v}}
  const float* o1w = (const float*)d_in[1];  const float* o1b = (const float*)d_in[2];
  const float* o1g = (const float*)d_in[3];  const float* o1be = (const float*)d_in[4];
  const float* o1m = (const float*)d_in[5];  const float* o1v = (const float*)d_in[6];
  const float* o2w = (const float*)d_in[7];  const float* o2b = (const float*)d_in[8];
  const float* o2g = (const float*)d_in[9];  const float* o2be = (const float*)d_in[10];
  const float* o2m = (const float*)d_in[11]; const float* o2v = (const float*)d_in[12];
  const float* o3w = (const float*)d_in[13]; const float* o3b = (const float*)d_in[14];
  const float* o3g = (const float*)d_in[15]; const float* o3be = (const float*)d_in[16];
  const float* o3m = (const float*)d_in[17]; const float* o3v = (const float*)d_in[18];
  const float* obw = (const float*)d_in[19]; const float* obb = (const float*)d_in[20];
  const float* obg = (const float*)d_in[21]; const float* obbe = (const float*)d_in[22];
  const float* obm = (const float*)d_in[23]; const float* obv = (const float*)d_in[24];
  const float* dfw = (const float*)d_in[25];
  const float* dcw = (const float*)d_in[26]; const float* dcb = (const float*)d_in[27];
  const float* dcg = (const float*)d_in[28]; const float* dcbe = (const float*)d_in[29];
  const float* dcm = (const float*)d_in[30]; const float* dcv = (const float*)d_in[31];
  const float* ncw = (const float*)d_in[32]; const float* ncb = (const float*)d_in[33];
  const float* ncg = (const float*)d_in[34]; const float* ncbe = (const float*)d_in[35];
  const float* ncm = (const float*)d_in[36]; const float* ncv = (const float*)d_in[37];
  const float* nsw = (const float*)d_in[38]; const float* nsb = (const float*)d_in[39];
  const float* nsg = (const float*)d_in[40]; const float* nsbe = (const float*)d_in[41];
  const float* nsm = (const float*)d_in[42]; const float* nsv = (const float*)d_in[43];
  const float* eew = (const float*)d_in[44]; const float* eeb = (const float*)d_in[45];
  const float* eeg = (const float*)d_in[46]; const float* eebe = (const float*)d_in[47];
  const float* eem = (const float*)d_in[48]; const float* eev = (const float*)d_in[49];

  char* ws = (char*)d_ws;
  const size_t NE = (size_t)B_ * C_ * HW_;     // 8,388,608 elements
  size_t off = 0;
  bf16_t* osum = (bf16_t*)(ws + off); off += NE * 2;            // NHWC bf16, 16MB
  bf16_t* dout = (bf16_t*)(ws + off); off += NE * 2;            // NHWC bf16
  bf16_t* xdcb = (bf16_t*)(ws + off); off += NE * 2;            // NHWC bf16
  bf16_t* xns  = (bf16_t*)(ws + off); off += NE * 2;            // NHWC bf16
  bf16_t* diff = (bf16_t*)(ws + off); off += NE * 2;            // NHWC bf16
  bf16_t* xnh  = (bf16_t*)(ws + off); off += NE * 2;            // NHWC bf16 copy of x
  float*  offs = (float*)(ws + off);  off += (size_t)B_ * HW_ * 18 * 4; // NHWC f32
  float*  xcon = (float*)(ws + off);  off += NE * 4;            // NCHW f32
  bf16_t* apk_off = (bf16_t*)(ws + off); off += (size_t)32 * 128 * 2;
  bf16_t* apk_dcb = (bf16_t*)(ws + off); off += (size_t)128 * 128 * 2;
  bf16_t* apk_ns  = (bf16_t*)(ws + off); off += (size_t)128 * 128 * 2;
  bf16_t* apk_ee  = (bf16_t*)(ws + off); off += (size_t)128 * 256 * 2;
  off = (off + 255) & ~(size_t)255;
  float* scO = (float*)(ws + off); off += 32 * 4;
  float* shO = (float*)(ws + off); off += 32 * 4;
  off = (off + 255) & ~(size_t)255;
  float* scD = (float*)(ws + off); off += 128 * 4;
  float* shD = (float*)(ws + off); off += 128 * 4;
  float* scN = (float*)(ws + off); off += 128 * 4;
  float* shN = (float*)(ws + off); off += 128 * 4;
  float* scE = (float*)(ws + off); off += 128 * 4;
  float* shE = (float*)(ws + off); off += 128 * 4;

  // 1) Prep: pack weights (A-fragment bf16), fold BN, x -> NHWC bf16.
  pack_w_kernel<<<16, 256, 0, stream>>>(obw, apk_off, 18, 128, 32);
  pack_w_kernel<<<64, 256, 0, stream>>>(dcw, apk_dcb, 128, 128, 128);
  pack_w_kernel<<<64, 256, 0, stream>>>(nsw, apk_ns, 128, 128, 128);
  pack_w_kernel<<<128, 256, 0, stream>>>(eew, apk_ee, 128, 256, 128);
  bnfold_kernel<<<1, 128, 0, stream>>>(obb, obg, obbe, obm, obv, 18, 32, scO, shO);
  bnfold_kernel<<<1, 128, 0, stream>>>(dcb, dcg, dcbe, dcm, dcv, 128, 128, scD, shD);
  bnfold_kernel<<<1, 128, 0, stream>>>(nsb, nsg, nsbe, nsm, nsv, 128, 128, scN, shN);
  bnfold_kernel<<<1, 128, 0, stream>>>(eeb, eeg, eebe, eem, eev, 128, 128, scE, shE);
  to_nhwc_kernel<<<32768, 256, 0, stream>>>(x, xnh);

  // 2) Fused depthwise gates -> o_sum (NHWC bf16).
  gates_kernel<<<B_ * C_ * H_, W_, 0, stream>>>(
      x, o1w, o1b, o1g, o1be, o1m, o1v,
         o2w, o2b, o2g, o2be, o2m, o2v,
         o3w, o3b, o3g, o3be, o3m, o3v, osum);

  // 3) off_bal 1x1 (C->18) GEMM -> offsets (NHWC f32).
  gemm1x1_kernel<2, 4, 0><<<1024, 128, 0, stream>>>(
      apk_off, osum, osum, 128, 18, scO, shO, nullptr, offs, nullptr);

  // 4) Deformable sample + einsum -> dout (NHWC bf16).
  deform_kernel<<<32768, 256, 0, stream>>>(x, offs, dfw, dout);

  // 5) dcb_bal 1x1 (C->C) GEMM, epilogue *x -> xdcb (NHWC bf16).
  gemm1x1_kernel<8, 4, 1><<<1024, 128, 0, stream>>>(
      apk_dcb, dout, dout, 128, 128, scD, shD, xnh, nullptr, xdcb);

  // 6) 15x15 depthwise ns_ctx + BN -> xcon (NCHW f32), LDS-tiled.
  nsctx_kernel<<<32768, 256, 0, stream>>>(x, ncw, ncb, ncg, ncbe, ncm, ncv, xcon);

  // 7) diff = xcon - boxavg3x3(xcon) -> NHWC bf16.
  pooldiff_kernel<<<32768, 256, 0, stream>>>(xcon, diff);

  // 8) ns_conv 1x1 (C->C) GEMM, epilogue sigmoid*x -> xns (NHWC bf16).
  gemm1x1_kernel<8, 4, 2><<<1024, 128, 0, stream>>>(
      apk_ns, diff, diff, 128, 128, scN, shN, xnh, nullptr, xns);

  // 9) ee_bal 1x1 (2C->C) GEMM over concat(xdcb, xns) -> d_out (NCHW f32).
  gemm1x1_kernel<8, 8, 3><<<1024, 128, 0, stream>>>(
      apk_ee, xdcb, xns, 128, 128, scE, shE, nullptr, (float*)d_out, nullptr);
}
